// EquivariantTransformer_dpm_6777458393834
// MI455X (gfx1250) — matrix operand order repaired and checked by reading, not compiled, not run on hardware
//
#include <hip/hip_runtime.h>
#include <math.h>

// ---------------- problem constants ----------------
#define BB   2
#define TTT  4
#define NAT  4096
#define DD   64
#define NG   (BB*TTT)          // 8 structures
#define NN   (NG*NAT)          // 32768 nodes
#define DEG  16
#define EE   (NN*DEG)          // 524288 edges
#define NTILE_E (EE/16)        // 32768 edge tiles
#define NTILE_N (NN/16)        // 2048  node tiles

typedef __attribute__((ext_vector_type(16))) __bf16 v16bf;
typedef __attribute__((ext_vector_type(8)))  __bf16 v8bf;
typedef __attribute__((ext_vector_type(8)))  float  v8f;
typedef __attribute__((ext_vector_type(2)))  float  v2f;

// fast silu: single v_exp_f32 + v_rcp_f32 instead of IEEE div expansion
__device__ __forceinline__ float silu_f(float x) {
    return x * __builtin_amdgcn_rcpf(1.0f + __expf(-x));
}

// ---------------- kernel 0: zero agg + vec_acc ----------------
__global__ void zero_kernel(float* __restrict__ p, int count) {
    int i = blockIdx.x * 256 + threadIdx.x;
    if (i < count) p[i] = 0.0f;
}

// ---------------- kernel 1: prep small tensors ----------------
// h0[b,d] = Ws[0,d] + t_b*Wt[0,d]
// bias[b,d] = sum_k h0[b,k]*(Wmsg[k,d]+Wmsg[64+k,d])
// wm2 = bf16(Wmsg[128:192]); wu = bf16(Wupd)
__global__ void prep_kernel(const float* __restrict__ t,
                            const float* __restrict__ Ws,
                            const float* __restrict__ Wt,
                            const float* __restrict__ Wmsg,
                            const float* __restrict__ Wupd,
                            float* __restrict__ h0,
                            float* __restrict__ bias,
                            __bf16* __restrict__ wm2,
                            __bf16* __restrict__ wu) {
    int tid = threadIdx.x;
    if (tid < BB * DD) {
        int b = tid >> 6, d = tid & 63;
        float tb = t[b];
        h0[b * DD + d] = Ws[d] + tb * Wt[d];
        float acc = 0.0f;
        for (int k = 0; k < DD; ++k) {
            float hk = Ws[k] + tb * Wt[k];
            acc += hk * (Wmsg[k * DD + d] + Wmsg[(DD + k) * DD + d]);
        }
        bias[b * DD + d] = acc;
    }
    for (int i = tid; i < DD * DD; i += 256) {
        wm2[i] = (__bf16)Wmsg[2 * DD * DD + i];
        wu[i]  = (__bf16)Wupd[i];
    }
}

// ---------------- kernel 2: positions ----------------
__global__ void pos_kernel(const float* __restrict__ x,
                           const float* __restrict__ x1,
                           const float* __restrict__ vm,
                           float* __restrict__ pos) {
    int i = blockIdx.x * 256 + threadIdx.x;
    if (i >= NN) return;
    float m = vm[i];
    #pragma unroll
    for (int c = 0; c < 3; ++c)
        pos[i * 3 + c] = x[i * 3 + c] * m + x1[i * 3 + c] * (1.0f - m);
}

// ---------------- kernel 3: per-edge messages (WMMA) ----------------
// wave handles 16-edge tiles: geometry (f32), ea=silu(edge_attr@W_edge) via
// v_wmma_f32_16x16x4_f32, LDS transpose -> bf16 A frags,
// msg=silu(ea@Wm2 + bias_b) via v_wmma_f32_16x16x32_bf16,
// atomic segment-sum into agg, spill msg (bf16) and edge_vec (f32).
__global__ __launch_bounds__(256) void edge_msg_kernel(
    const float* __restrict__ pos, const float* __restrict__ cell,
    const int* __restrict__ edge_index, const int* __restrict__ to_jimages,
    const float* __restrict__ Wedge, const __bf16* __restrict__ Wm2,
    const float* __restrict__ bias, float* __restrict__ agg,
    __bf16* __restrict__ msgw, float* __restrict__ evecw) {
    __shared__ __align__(16) __bf16 lds_ea[8][16 * DD];
    const int lane = threadIdx.x & 31;
    const int wv = threadIdx.x >> 5;
    const int r  = lane & 15;      // row within tile / output col
    const int hi = lane >> 4;      // lane half
    const int gwave  = blockIdx.x * 8 + wv;
    const int nwaves = gridDim.x * 8;

    // B fragments of W_edge (f32 4x16 per ntile):
    // lane col = r, VGPR j -> K = 2*hi + j
    v2f bw[4];
    #pragma unroll
    for (int t2 = 0; t2 < 4; ++t2) {
        int col = t2 * 16 + r;
        bw[t2].x = Wedge[(2 * hi + 0) * DD + col];
        bw[t2].y = Wedge[(2 * hi + 1) * DD + col];
    }
    // B fragments of Wm2 (bf16 32x16): half h<8 -> K = kc*32 + 8*hi + h; h>=8 -> +8
    v16bf bm[2][4];
    #pragma unroll
    for (int kc = 0; kc < 2; ++kc)
        #pragma unroll
        for (int t2 = 0; t2 < 4; ++t2) {
            int col = t2 * 16 + r;
            #pragma unroll
            for (int h = 0; h < 16; ++h)
                bm[kc][t2][h] = Wm2[(kc * 32 + 8 * hi + h + (h >= 8 ? 8 : 0)) * DD + col];
        }

    for (int tile = gwave; tile < NTILE_E; tile += nwaves) {
        // both lane-halves compute edge (tile*16 + r) redundantly
        int e   = tile * 16 + r;
        int src = edge_index[e];
        int dst = edge_index[EE + e];
        int g   = src >> 12;                          // structure id
        float tj0 = (float)to_jimages[e * 3 + 0];
        float tj1 = (float)to_jimages[e * 3 + 1];
        float tj2 = (float)to_jimages[e * 3 + 2];
        const float* C = cell + g * 9;
        float ox = tj0 * C[0] + tj1 * C[3] + tj2 * C[6];
        float oy = tj0 * C[1] + tj1 * C[4] + tj2 * C[7];
        float oz = tj0 * C[2] + tj1 * C[5] + tj2 * C[8];
        float ex = pos[dst * 3 + 0] - pos[src * 3 + 0] + ox;
        float ey = pos[dst * 3 + 1] - pos[src * 3 + 1] + oy;
        float ez = pos[dst * 3 + 2] - pos[src * 3 + 2] + oz;
        float el = sqrtf(ex * ex + ey * ey + ez * ez + 1e-12f);
        if (hi == 0) {
            evecw[e * 3 + 0] = ex;
            evecw[e * 3 + 1] = ey;
            evecw[e * 3 + 2] = ez;
        }
        // A frag (f32 16x4): lanes 0-15 hold K=0,1; lanes 16-31 hold K=2,3
        v2f a;
        a.x = hi ? ez : ex;
        a.y = hi ? el : ey;

        __bf16* L = lds_ea[wv];
        #pragma unroll
        for (int t2 = 0; t2 < 4; ++t2) {
            v8f acc = {};
            acc = __builtin_amdgcn_wmma_f32_16x16x4_f32(
                false, a, false, bw[t2], (short)0, acc, false, false);
            #pragma unroll
            for (int f = 0; f < 8; ++f) {
                float v = silu_f(acc[f]);
                L[(f + 8 * hi) * DD + t2 * 16 + r] = (__bf16)v; // row-major ea
            }
        }
        __builtin_amdgcn_wave_barrier();
        asm volatile("s_wait_dscnt 0" ::: "memory");

        // A frags (bf16 16x32): row=r, half h<8 -> K=kc*32+8*hi+h; h>=8 -> +16
        v16bf afr[2];
        #pragma unroll
        for (int kc = 0; kc < 2; ++kc) {
            const __bf16* rp = &L[r * DD + kc * 32 + hi * 8];
            v8bf lo  = *(const v8bf*)(rp);
            v8bf hi8 = *(const v8bf*)(rp + 16);
            #pragma unroll
            for (int h = 0; h < 8; ++h) {
                afr[kc][h]     = lo[h];
                afr[kc][8 + h] = hi8[h];
            }
        }
        __builtin_amdgcn_wave_barrier();

        int b = tile >> 14;   // 16384 tiles per batch index
        #pragma unroll
        for (int t2 = 0; t2 < 4; ++t2) {
            v8f acc = {};
            acc = __builtin_amdgcn_wmma_f32_16x16x32_bf16(
                false, afr[0], false, bm[0][t2], (short)0, acc, false, false);
            acc = __builtin_amdgcn_wmma_f32_16x16x32_bf16(
                false, afr[1], false, bm[1][t2], (short)0, acc, false, false);
            int col = t2 * 16 + r;
            float bv = bias[b * DD + col];
            #pragma unroll
            for (int f = 0; f < 8; ++f) {
                float m = silu_f(acc[f] + bv);
                int M = f + 8 * hi;
                int dstM = __shfl(dst, f + 8 * hi, 32);
                atomicAdd(&agg[(size_t)dstM * DD + col], m);
                msgw[(size_t)(tile * 16 + M) * DD + col] = (__bf16)m;
            }
        }
    }
}

// ---------------- kernel 4: node update + gate (WMMA) ----------------
// h = h0_b + silu(agg @ W_upd); g = silu(h)*w_dec
__global__ __launch_bounds__(256) void node_kernel(
    const float* __restrict__ agg, const __bf16* __restrict__ Wu,
    const float* __restrict__ h0, const float* __restrict__ wdec,
    float* __restrict__ gout) {
    const int lane = threadIdx.x & 31;
    const int wv = threadIdx.x >> 5;
    const int r  = lane & 15;
    const int hi = lane >> 4;
    const int gwave  = blockIdx.x * 8 + wv;
    const int nwaves = gridDim.x * 8;

    v16bf bu[2][4];
    #pragma unroll
    for (int kc = 0; kc < 2; ++kc)
        #pragma unroll
        for (int t2 = 0; t2 < 4; ++t2) {
            int col = t2 * 16 + r;
            #pragma unroll
            for (int h = 0; h < 16; ++h)
                bu[kc][t2][h] = Wu[(kc * 32 + 8 * hi + h + (h >= 8 ? 8 : 0)) * DD + col];
        }

    for (int tile = gwave; tile < NTILE_N; tile += nwaves) {
        const float* rowp = agg + (size_t)(tile * 16 + r) * DD;
        v16bf afr[2];
        #pragma unroll
        for (int kc = 0; kc < 2; ++kc)
            #pragma unroll
            for (int h = 0; h < 8; ++h) {
                afr[kc][h]     = (__bf16)rowp[kc * 32 + hi * 8 + h];
                afr[kc][8 + h] = (__bf16)rowp[kc * 32 + hi * 8 + 16 + h];
            }
        int b = tile >> 10;   // 1024 tiles per batch index
        #pragma unroll
        for (int t2 = 0; t2 < 4; ++t2) {
            v8f acc = {};
            acc = __builtin_amdgcn_wmma_f32_16x16x32_bf16(
                false, afr[0], false, bu[0][t2], (short)0, acc, false, false);
            acc = __builtin_amdgcn_wmma_f32_16x16x32_bf16(
                false, afr[1], false, bu[1][t2], (short)0, acc, false, false);
            int col = t2 * 16 + r;
            float h0v = h0[b * DD + col];
            float wd  = wdec[col];
            #pragma unroll
            for (int f = 0; f < 8; ++f) {
                float hv = h0v + silu_f(acc[f]);
                float gv = silu_f(hv) * wd;
                gout[(size_t)(tile * 16 + f + 8 * hi) * DD + col] = gv;
            }
        }
    }
}

// ---------------- kernel 5: edge decode ----------------
// s_e = msg[e]·g[dst];  vec_acc[dst,c] += edge_vec[e,c]*s_e
__global__ __launch_bounds__(256) void decode_kernel(
    const int* __restrict__ edge_index, const __bf16* __restrict__ msgw,
    const float* __restrict__ g, const float* __restrict__ evecw,
    float* __restrict__ vacc) {
    int e = blockIdx.x * 256 + threadIdx.x;
    if (e >= EE) return;
    int dst = edge_index[EE + e];
    const __bf16* mp = msgw + (size_t)e * DD;
    const float*  gp = g + (size_t)dst * DD;
    float s = 0.0f;
    #pragma unroll
    for (int k = 0; k < DD; k += 8) {
        v8bf mv = *(const v8bf*)(mp + k);
        float4 g0 = *(const float4*)(gp + k);
        float4 g1 = *(const float4*)(gp + k + 4);
        s += (float)mv[0] * g0.x + (float)mv[1] * g0.y +
             (float)mv[2] * g0.z + (float)mv[3] * g0.w +
             (float)mv[4] * g1.x + (float)mv[5] * g1.y +
             (float)mv[6] * g1.z + (float)mv[7] * g1.w;
    }
    atomicAdd(&vacc[(size_t)dst * 3 + 0], evecw[e * 3 + 0] * s);
    atomicAdd(&vacc[(size_t)dst * 3 + 1], evecw[e * 3 + 1] * s);
    atomicAdd(&vacc[(size_t)dst * 3 + 2], evecw[e * 3 + 2] * s);
}

// ---------------- kernel 6: finalize (mask) ----------------
__global__ void final_kernel(const float* __restrict__ vacc,
                             const float* __restrict__ vm,
                             float* __restrict__ out) {
    int i = blockIdx.x * 256 + threadIdx.x;
    if (i >= NN * 3) return;
    out[i] = vacc[i] * vm[i / 3];
}

// ---------------- workspace layout (bytes) ----------------
#define OFF_POS   ((size_t)0)                         // NN*3 f32
#define OFF_AGG   (OFF_POS  + (size_t)NN*3*4)         // NN*64 f32
#define OFF_VEC   (OFF_AGG  + (size_t)NN*DD*4)        // NN*3 f32
#define OFF_G     (OFF_VEC  + (size_t)NN*3*4)         // NN*64 f32
#define OFF_H0    (OFF_G    + (size_t)NN*DD*4)        // 2*64 f32
#define OFF_BIAS  (OFF_H0   + (size_t)BB*DD*4)        // 2*64 f32
#define OFF_WM2   (OFF_BIAS + (size_t)BB*DD*4)        // 64*64 bf16
#define OFF_WUPD  (OFF_WM2  + (size_t)DD*DD*2)        // 64*64 bf16
#define OFF_EVEC  (OFF_WUPD + (size_t)DD*DD*2)        // E*3 f32
#define OFF_MSG   (OFF_EVEC + (size_t)EE*3*4)         // E*64 bf16

extern "C" void kernel_launch(void* const* d_in, const int* in_sizes, int n_in,
                              void* d_out, int out_size, void* d_ws, size_t ws_size,
                              hipStream_t stream) {
    const float* x    = (const float*)d_in[0];
    const float* t    = (const float*)d_in[1];
    const float* cell = (const float*)d_in[2];
    const float* x1   = (const float*)d_in[3];
    const float* vm   = (const float*)d_in[4];
    const float* Ws   = (const float*)d_in[5];
    const float* Wt   = (const float*)d_in[6];
    const float* We   = (const float*)d_in[7];
    const float* Wm   = (const float*)d_in[8];
    const float* Wu   = (const float*)d_in[9];
    const float* wdec = (const float*)d_in[10];
    const int* eidx   = (const int*)d_in[11];
    const int* tji    = (const int*)d_in[12];
    float* out = (float*)d_out;

    char* ws = (char*)d_ws;
    float*  pos    = (float*)(ws + OFF_POS);
    float*  agg    = (float*)(ws + OFF_AGG);
    float*  vacc   = (float*)(ws + OFF_VEC);
    float*  gbuf   = (float*)(ws + OFF_G);
    float*  h0     = (float*)(ws + OFF_H0);
    float*  bias   = (float*)(ws + OFF_BIAS);
    __bf16* wm2    = (__bf16*)(ws + OFF_WM2);
    __bf16* wupd   = (__bf16*)(ws + OFF_WUPD);
    float*  evecw  = (float*)(ws + OFF_EVEC);
    __bf16* msgw   = (__bf16*)(ws + OFF_MSG);

    // agg and vacc are contiguous: zero both in one pass
    int zcount = NN * DD + NN * 3;
    zero_kernel<<<(zcount + 255) / 256, 256, 0, stream>>>(agg, zcount);
    prep_kernel<<<1, 256, 0, stream>>>(t, Ws, Wt, Wm, Wu, h0, bias, wm2, wupd);
    pos_kernel<<<(NN + 255) / 256, 256, 0, stream>>>(x, x1, vm, pos);
    edge_msg_kernel<<<512, 256, 0, stream>>>(pos, cell, eidx, tji, We, wm2,
                                             bias, agg, msgw, evecw);
    node_kernel<<<64, 256, 0, stream>>>(agg, wupd, h0, wdec, gbuf);
    decode_kernel<<<EE / 256, 256, 0, stream>>>(eidx, msgw, gbuf, evecw, vacc);
    final_kernel<<<(NN * 3 + 255) / 256, 256, 0, stream>>>(vacc, vm, out);
}
